// SimplePoseTAG_14516989461135
// MI455X (gfx1250) — matrix operand order repaired and checked by reading, not compiled, not run on hardware
//
#include <hip/hip_runtime.h>

// ---------------------------------------------------------------------------
// SimplePoseTAG for MI455X (gfx1250): TAGConv GNN
//   - CSR built on device (deg count -> single-block scan -> fill)
//   - tagconv linear done per-hop as WMMA bf16 GEMM accumulation into fp32 C
//   - v_wmma_f32_16x16x32_bf16; block = 128x128 tile, wave = 16x128 strip
//     (8 accumulator frags / wave, 32 WMMAs / wave)
// ---------------------------------------------------------------------------

#define NN   50000
#define EE   600000
#define HH   128
#define KHOP 5
#define CC   60
#define NMODC 12

typedef __attribute__((ext_vector_type(16))) __bf16 v16bf;
typedef __attribute__((ext_vector_type(8)))  __bf16 v8bf;
typedef __attribute__((ext_vector_type(8)))  float  v8f;

static __device__ __forceinline__ unsigned short f2bf_bits(float f) {
    union { float f; unsigned u; } v; v.f = f;
    unsigned r = v.u + 0x7FFFu + ((v.u >> 16) & 1u);   // round-to-nearest-even
    return (unsigned short)(r >> 16);
}

// ------------------------------- small utils -------------------------------

__global__ void k_zero_f(float* p, long n) {
    long i = (long)blockIdx.x * blockDim.x + threadIdx.x;
    if (i < n) p[i] = 0.0f;
}
__global__ void k_zero_i(int* p, long n) {
    long i = (long)blockIdx.x * blockDim.x + threadIdx.x;
    if (i < n) p[i] = 0;
}
// convert fp32 weights to bf16, transposing each 128x128 block: dst[b][n][k] = src[b][k][n]
__global__ void k_cvt_bf16_t(const float* __restrict__ src, unsigned short* __restrict__ dst, long nElems) {
    long i = (long)blockIdx.x * blockDim.x + threadIdx.x;
    if (i >= nElems) return;
    long blk = i >> 14;                 // /16384
    int  on  = (int)(i & 16383);
    int  n   = on >> 7, k = on & 127;
    dst[i] = f2bf_bits(src[(blk << 14) + (long)k * HH + n]);
}

// ------------------------------- graph setup -------------------------------

__global__ void k_count_deg(const int* __restrict__ dst, int* __restrict__ deg, int e) {
    int i = blockIdx.x * blockDim.x + threadIdx.x;
    if (i < e) atomicAdd(&deg[dst[i]], 1);
}
__global__ void k_norm(const int* __restrict__ deg, float* __restrict__ norm, int n) {
    int i = blockIdx.x * blockDim.x + threadIdx.x;
    if (i < n) {
        int d = deg[i]; if (d < 1) d = 1;
        norm[i] = rsqrtf((float)d);
    }
}
// exclusive prefix sum over deg[n] -> row_off[n+1], single 1024-thread block
__global__ void k_scan(const int* __restrict__ deg, int* __restrict__ row_off, int n) {
    __shared__ int buf[1024];
    __shared__ int carry;
    if (threadIdx.x == 0) carry = 0;
    __syncthreads();
    for (int base = 0; base < n; base += 1024) {
        int i = base + (int)threadIdx.x;
        int v = (i < n) ? deg[i] : 0;
        buf[threadIdx.x] = v;
        __syncthreads();
        for (int off = 1; off < 1024; off <<= 1) {
            int t = (threadIdx.x >= (unsigned)off) ? buf[threadIdx.x - off] : 0;
            __syncthreads();
            buf[threadIdx.x] += t;
            __syncthreads();
        }
        int excl = buf[threadIdx.x] - v;
        if (i < n) row_off[i] = carry + excl;
        __syncthreads();
        if (threadIdx.x == 1023) carry += buf[1023];
        __syncthreads();
    }
    if (threadIdx.x == 0) row_off[n] = carry;
}
__global__ void k_csr_fill(const int* __restrict__ src, const int* __restrict__ dst,
                           const int* __restrict__ row_off, int* __restrict__ cursor,
                           int* __restrict__ csr_src, int e) {
    int i = blockIdx.x * blockDim.x + threadIdx.x;
    if (i < e) {
        int d = dst[i];
        int pos = atomicAdd(&cursor[d], 1);
        csr_src[row_off[d] + pos] = src[i];
    }
}

// edge embedding + scatter-add onto dst (one time)
__global__ void k_edge_agg(const float* __restrict__ ef, const float* __restrict__ ew,
                           const float* __restrict__ eb, const int* __restrict__ dst,
                           float* __restrict__ agg, int e) {
    long idx = (long)blockIdx.x * blockDim.x + threadIdx.x;
    if (idx >= (long)e * HH) return;
    int ee = (int)(idx >> 7), j = (int)(idx & 127);
    float v = eb[j];
#pragma unroll
    for (int q = 0; q < 4; ++q) v += ef[ee * 4 + q] * ew[q * HH + j];
    atomicAdd(&agg[(long)dst[ee] * HH + j], v);
}

__global__ void k_node_init(const float* __restrict__ nf, const float* __restrict__ pe,
                            const float* __restrict__ in_w, const float* __restrict__ in_b,
                            const float* __restrict__ pos_w, const float* __restrict__ pos_b,
                            const float* __restrict__ agg, float* __restrict__ h, int n) {
    long idx = (long)blockIdx.x * blockDim.x + threadIdx.x;
    if (idx >= (long)n * HH) return;
    int i = (int)(idx >> 7), j = (int)(idx & 127);
    float v = in_b[j] + pos_b[j] + agg[idx];
    v += nf[i * 2 + 0] * in_w[j] + nf[i * 2 + 1] * in_w[HH + j];
#pragma unroll
    for (int q = 0; q < KHOP; ++q) v += pe[i * KHOP + q] * pos_w[q * HH + j];
    h[idx] = v;
}

// one hop: out[i,:] = norm[i] * sum_{s in in(i)} norm[s] * cur[s,:]
__global__ void k_propagate(const float* __restrict__ cur, float* __restrict__ out,
                            const float* __restrict__ norm, const int* __restrict__ row_off,
                            const int* __restrict__ csr_src) {
    int node = blockIdx.x * 2 + (int)(threadIdx.x >> 7);
    int j = (int)(threadIdx.x & 127);
    int beg = row_off[node], end = row_off[node + 1];
    float acc = 0.0f;
    for (int ep = beg; ep < end; ++ep) {
        int s = csr_src[ep];
        acc += norm[s] * cur[(long)s * HH + j];
    }
    out[(long)node * HH + j] = norm[node] * acc;
}

// ----------------------- WMMA bf16 GEMM: C[M,128] ops -----------------------
// C = A[M,128] * W[128,128] (+bias | +=C) (+residual) (relu?)
// Wt is the 128x128 weight block pre-transposed to [n][k] in bf16.
// Block covers 128 rows x 128 cols; each wave owns a 16-row strip and all
// 8 column tiles (8 v8f accumulators, 32 chained WMMAs).
__global__ void __launch_bounds__(256)
k_gemm128_wmma(const float* __restrict__ A, const unsigned short* __restrict__ Wt,
               const float* __restrict__ bias, const float* __restrict__ residual,
               float* __restrict__ C, int M, int accumFromC, int reluStore) {
    __shared__ unsigned short sA[128][136];   // bf16 bits; 272B row stride (16B aligned, 4-bank rotate)
    __shared__ unsigned short sBt[128][136];  // Wt: [n][k]

    const int tid  = (int)threadIdx.x;
    const int lane = tid & 31;
    const int wave = tid >> 5;
    const int m0   = blockIdx.x * 128;

    // stage A tile (fp32 -> bf16), 8 elems (16B) per thread-iteration
    for (int cidx = tid; cidx < 128 * 16; cidx += 256) {
        int r = cidx >> 4, k0 = (cidx & 15) * 8;
        int row = m0 + r;
        unsigned u[4];
        if (row < M) {
            const float* ap = A + (long)row * HH + k0;
#pragma unroll
            for (int q = 0; q < 4; ++q)
                u[q] = (unsigned)f2bf_bits(ap[2 * q]) | ((unsigned)f2bf_bits(ap[2 * q + 1]) << 16);
        } else {
            u[0] = u[1] = u[2] = u[3] = 0;
        }
        uint4 val; val.x = u[0]; val.y = u[1]; val.z = u[2]; val.w = u[3];
        *reinterpret_cast<uint4*>(&sA[r][k0]) = val;
    }
    // stage W (already bf16 + transposed): straight 16B copies
    for (int cidx = tid; cidx < 128 * 16; cidx += 256) {
        int n = cidx >> 4, k0 = (cidx & 15) * 8;
        uint4 v = *reinterpret_cast<const uint4*>(Wt + n * HH + k0);
        *reinterpret_cast<uint4*>(&sBt[n][k0]) = v;
    }
    __syncthreads();

    const int hi   = lane >> 4;        // lane half selects K sub-block
    const int lr   = lane & 15;
    const int mloc = wave * 16 + lr;   // local A row for this lane

    v8f c[8];
    if (accumFromC) {
#pragma unroll
        for (int n = 0; n < 8; ++n)
#pragma unroll
            for (int r = 0; r < 8; ++r) {
                long row = m0 + wave * 16 + hi * 8 + r;
                c[n][r] = (row < M) ? C[row * HH + n * 16 + lr] : 0.0f;
            }
    } else {
#pragma unroll
        for (int n = 0; n < 8; ++n) {
            float bv = bias[n * 16 + lr];
#pragma unroll
            for (int r = 0; r < 8; ++r) c[n][r] = bv;
        }
    }

#pragma unroll
    for (int kc = 0; kc < 4; ++kc) {
        // A frag: lanes 0-15 hold K {0..7,16..23}, lanes 16-31 hold K {8..15,24..31}
        int ka = kc * 32 + hi * 8;
        v8bf alo = *reinterpret_cast<const v8bf*>(&sA[mloc][ka]);
        v8bf ahi = *reinterpret_cast<const v8bf*>(&sA[mloc][ka + 16]);
        v16bf a = __builtin_shufflevector(alo, ahi, 0,1,2,3,4,5,6,7,8,9,10,11,12,13,14,15);
        int kb = kc * 32 + hi * 8;
#pragma unroll
        for (int n = 0; n < 8; ++n) {
            // B frag: lane = column; lanes 0-15 K {0..7,16..23}, lanes 16-31 K {8..15,24..31}
            v8bf blo = *reinterpret_cast<const v8bf*>(&sBt[n * 16 + lr][kb]);
            v8bf bhi = *reinterpret_cast<const v8bf*>(&sBt[n * 16 + lr][kb + 16]);
            v16bf b = __builtin_shufflevector(blo, bhi, 0,1,2,3,4,5,6,7,8,9,10,11,12,13,14,15);
            c[n] = __builtin_amdgcn_wmma_f32_16x16x32_bf16(false, a, false, b, (short)0, c[n], false, false);
        }
    }

#pragma unroll
    for (int n = 0; n < 8; ++n)
#pragma unroll
        for (int r = 0; r < 8; ++r) {
            long row = m0 + wave * 16 + hi * 8 + r;
            if (row < M) {
                float v = c[n][r];
                if (residual) v += residual[row * HH + n * 16 + lr];
                if (reluStore) v = fmaxf(v, 0.0f);
                C[row * HH + n * 16 + lr] = v;
            }
        }
}

// ------------------------------- batch norm --------------------------------

__global__ void k_col_stats(const float* __restrict__ X, float* __restrict__ sum,
                            float* __restrict__ sumsq, int M) {
    int j = (int)(threadIdx.x & 127);
    int half = (int)(threadIdx.x >> 7);
    int r0 = blockIdx.x * 128 + half;
    int rend = blockIdx.x * 128 + 128; if (rend > M) rend = M;
    float s = 0.0f, q = 0.0f;
    for (int r = r0; r < rend; r += 2) {
        float v = X[(long)r * HH + j];
        s += v; q += v * v;
    }
    __shared__ float ls[256], lq[256];
    ls[threadIdx.x] = s; lq[threadIdx.x] = q;
    __syncthreads();
    if (half == 0) {
        atomicAdd(&sum[j],   ls[j] + ls[j + 128]);
        atomicAdd(&sumsq[j], lq[j] + lq[j + 128]);
    }
}
__global__ void k_bn_finalize(const float* __restrict__ sum, const float* __restrict__ sumsq,
                              const float* __restrict__ g, const float* __restrict__ b,
                              float* __restrict__ scale, float* __restrict__ shift, int M) {
    int j = (int)threadIdx.x;
    float mean = sum[j] / (float)M;
    float var  = sumsq[j] / (float)M - mean * mean;
    float sc   = rsqrtf(var + 1e-5f) * g[j];
    scale[j] = sc;
    shift[j] = b[j] - mean * sc;
}
__global__ void k_bn_apply_relu(const float* __restrict__ X, const float* __restrict__ scale,
                                const float* __restrict__ shift, float* __restrict__ Y, int M) {
    long idx = (long)blockIdx.x * blockDim.x + threadIdx.x;
    if (idx >= (long)M * HH) return;
    int j = (int)(idx & 127);
    Y[idx] = fmaxf(X[idx] * scale[j] + shift[j], 0.0f);
}

// --------------------------------- heads -----------------------------------

__global__ void k_pose_head(const float* __restrict__ T, const float* __restrict__ w2,
                            const float* __restrict__ b2, float* __restrict__ out, int M) {
    int wave = (int)(threadIdx.x >> 5), lane = (int)(threadIdx.x & 31);
    int i = blockIdx.x * 8 + wave;
    if (i >= M) return;
    float a0 = 0.0f, a1 = 0.0f, a2 = 0.0f;
    for (int j = lane; j < HH; j += 32) {
        float h = T[(long)i * HH + j];
        a0 += h * w2[j * 3 + 0];
        a1 += h * w2[j * 3 + 1];
        a2 += h * w2[j * 3 + 2];
    }
    for (int off = 16; off > 0; off >>= 1) {
        a0 += __shfl_down(a0, off);
        a1 += __shfl_down(a1, off);
        a2 += __shfl_down(a2, off);
    }
    if (lane == 0) {
        out[(long)i * 3 + 0] = a0 + b2[0];
        out[(long)i * 3 + 1] = a1 + b2[1];
        out[(long)i * 3 + 2] = a2 + b2[2];
    }
}
__global__ void k_label_head(const float* __restrict__ colsum, const float* __restrict__ w1,
                             const float* __restrict__ b1, const float* __restrict__ w2,
                             const float* __restrict__ b2, float* __restrict__ out, int M) {
    __shared__ float y[HH], hid[HH];
    int j = (int)threadIdx.x;
    y[j] = colsum[j] / (float)M;
    __syncthreads();
    float a = b1[j];
    for (int k = 0; k < HH; ++k) a += y[k] * w1[k * HH + j];
    hid[j] = fmaxf(a, 0.0f);
    __syncthreads();
    if (j < CC) {
        float v = b2[j];
        for (int k = 0; k < HH; ++k) v += hid[k] * w2[k * CC + j];
        out[j] = v;
    }
}

// --------------------------------- driver ----------------------------------

static inline size_t alignup(size_t x) { return (x + 255) & ~(size_t)255; }

extern "C" void kernel_launch(void* const* d_in, const int* in_sizes, int n_in,
                              void* d_out, int out_size, void* d_ws, size_t ws_size,
                              hipStream_t stream) {
    (void)in_sizes; (void)n_in; (void)out_size; (void)ws_size;
    const float* nf      = (const float*)d_in[0];
    const float* pe      = (const float*)d_in[1];
    const float* ef      = (const float*)d_in[2];
    const int*   src     = (const int*)d_in[3];
    const int*   dst     = (const int*)d_in[4];
    const float* in_w    = (const float*)d_in[5];
    const float* in_b    = (const float*)d_in[6];
    const float* pos_w   = (const float*)d_in[7];
    const float* pos_b   = (const float*)d_in[8];
    const float* edge_w  = (const float*)d_in[9];
    const float* edge_b  = (const float*)d_in[10];
    const float* conv_w  = (const float*)d_in[11];
    const float* conv_b  = (const float*)d_in[12];
    const float* bn_g    = (const float*)d_in[13];
    const float* bn_b    = (const float*)d_in[14];
    const float* ff_w    = (const float*)d_in[15];
    const float* ff_b    = (const float*)d_in[16];
    const float* pose_w1 = (const float*)d_in[17];
    const float* pose_b1 = (const float*)d_in[18];
    const float* pose_w2 = (const float*)d_in[19];
    const float* pose_b2 = (const float*)d_in[20];
    const float* lab_w1  = (const float*)d_in[21];
    const float* lab_b1  = (const float*)d_in[22];
    const float* lab_w2  = (const float*)d_in[23];
    const float* lab_b2  = (const float*)d_in[24];
    float* out = (float*)d_out;

    // workspace carve-up
    char* base = (char*)d_ws;
    size_t off = 0;
    auto take = [&](size_t bytes) { char* p = base + off; off += alignup(bytes); return p; };
    float* norm    = (float*)take(sizeof(float) * NN);
    int*   deg     = (int*)  take(sizeof(int) * NN);
    int*   row_off = (int*)  take(sizeof(int) * (NN + 1));
    int*   cursor  = (int*)  take(sizeof(int) * NN);
    int*   csr_src = (int*)  take(sizeof(int) * EE);
    float* colsum  = (float*)take(sizeof(float) * 4 * HH);  // sum | sumsq | scale | shift
    float* colsumsq = colsum + HH;
    float* bscale   = colsum + 2 * HH;
    float* bshift   = colsum + 3 * HH;
    const long convElems = (long)NMODC * 2 * (KHOP + 1) * HH * HH;  // 2,359,296
    const long ffElems   = (long)NMODC * HH * HH;                   //   196,608
    const long poseElems = (long)HH * HH;                           //    16,384
    unsigned short* Wbf = (unsigned short*)take(sizeof(unsigned short) * (convElems + ffElems + poseElems));
    unsigned short* Wbf_ff   = Wbf + convElems;
    unsigned short* Wbf_pose = Wbf + convElems + ffElems;
    float* Hbuf = (float*)take(sizeof(float) * (long)NN * HH);
    float* t1   = (float*)take(sizeof(float) * (long)NN * HH);
    float* P1   = (float*)take(sizeof(float) * (long)NN * HH);
    float* P2   = (float*)take(sizeof(float) * (long)NN * HH);
    float* tmpG = (float*)take(sizeof(float) * (long)NN * HH);

    const long NHtot = (long)NN * HH;
    const int TB = 256;
    dim3 blk(TB);

    // ---- graph setup (per call; buffers are poisoned/dirty otherwise) ----
    k_zero_i<<<dim3((NN + TB - 1) / TB), blk, 0, stream>>>(deg, NN);
    k_count_deg<<<dim3((EE + TB - 1) / TB), blk, 0, stream>>>(dst, deg, EE);
    k_norm<<<dim3((NN + TB - 1) / TB), blk, 0, stream>>>(deg, norm, NN);
    k_scan<<<dim3(1), dim3(1024), 0, stream>>>(deg, row_off, NN);
    k_zero_i<<<dim3((NN + TB - 1) / TB), blk, 0, stream>>>(cursor, NN);
    k_csr_fill<<<dim3((EE + TB - 1) / TB), blk, 0, stream>>>(src, dst, row_off, cursor, csr_src, EE);

    // ---- pre-convert GEMM weights to bf16, transposed per 128x128 block ----
    k_cvt_bf16_t<<<dim3((int)((convElems + TB - 1) / TB)), blk, 0, stream>>>(conv_w, Wbf, convElems);
    k_cvt_bf16_t<<<dim3((int)((ffElems + TB - 1) / TB)), blk, 0, stream>>>(ff_w, Wbf_ff, ffElems);
    k_cvt_bf16_t<<<dim3((int)((poseElems + TB - 1) / TB)), blk, 0, stream>>>(pose_w1, Wbf_pose, poseElems);

    // ---- input embedding ----
    k_zero_f<<<dim3((int)((NHtot + TB - 1) / TB)), blk, 0, stream>>>(t1, NHtot);
    k_edge_agg<<<dim3((int)(((long)EE * HH + TB - 1) / TB)), blk, 0, stream>>>(ef, edge_w, edge_b, dst, t1, EE);
    k_node_init<<<dim3((int)((NHtot + TB - 1) / TB)), blk, 0, stream>>>(nf, pe, in_w, in_b, pos_w, pos_b, t1, Hbuf, NN);

    const dim3 gGemm((NN + 127) / 128); // 391 blocks, 8 waves, 8 c-frags/wave
    const dim3 gProp(NN / 2);           // 2 nodes per 256-thread block
    const dim3 gElem((int)((NHtot + TB - 1) / TB));
    const dim3 gStat((NN + 127) / 128);

    auto tagconv = [&](const float* X, float* Y, int m, int s) {
        const unsigned short* Wb0 = Wbf + ((long)(m * 2 + s) * (KHOP + 1) * HH) * HH;
        const float* cb = conv_b + (m * 2 + s) * HH;
        // hop 0: tmpG = X @ W0 + bias
        k_gemm128_wmma<<<gGemm, blk, 0, stream>>>(X, Wb0, cb, nullptr, tmpG, NN, 0, 0);
        const float* cur = X;
        for (int k = 1; k <= KHOP; ++k) {
            float* nxt = (k & 1) ? P1 : P2;
            k_propagate<<<gProp, blk, 0, stream>>>(cur, nxt, norm, row_off, csr_src);
            k_gemm128_wmma<<<gGemm, blk, 0, stream>>>(nxt, Wb0 + (long)k * HH * HH, nullptr, nullptr, tmpG, NN, 1, 0);
            cur = nxt;
        }
        // BN (training stats) + ReLU
        k_zero_f<<<dim3(1), blk, 0, stream>>>(colsum, 2 * HH);
        k_col_stats<<<gStat, blk, 0, stream>>>(tmpG, colsum, colsumsq, NN);
        k_bn_finalize<<<dim3(1), dim3(HH), 0, stream>>>(colsum, colsumsq,
                                                        bn_g + (m * 2 + s) * HH, bn_b + (m * 2 + s) * HH,
                                                        bscale, bshift, NN);
        k_bn_apply_relu<<<gElem, blk, 0, stream>>>(tmpG, bscale, bshift, Y, NN);
    };

    for (int m = 0; m < NMODC; ++m) {
        tagconv(Hbuf, t1, m, 0);
        tagconv(t1,   t1, m, 1);
        // h = t1 @ ff_w[m] + ff_b[m] + h_in  (residual from Hbuf, written in place)
        k_gemm128_wmma<<<gGemm, blk, 0, stream>>>(t1, Wbf_ff + (long)m * HH * HH,
                                                  ff_b + m * HH, Hbuf, Hbuf, NN, 0, 0);
    }

    // pose head: t1 = relu(h @ pose_w1 + pose_b1); pose = t1 @ pose_w2 + pose_b2
    k_gemm128_wmma<<<gGemm, blk, 0, stream>>>(Hbuf, Wbf_pose, pose_b1, nullptr, t1, NN, 0, 1);
    k_pose_head<<<dim3((NN + 7) / 8), blk, 0, stream>>>(t1, pose_w2, pose_b2, out, NN);

    // label head: y = mean(h, axis=0); labels = relu(y@W1+b1)@W2+b2
    k_zero_f<<<dim3(1), blk, 0, stream>>>(colsum, 2 * HH);
    k_col_stats<<<gStat, blk, 0, stream>>>(Hbuf, colsum, colsumsq, NN);
    k_label_head<<<dim3(1), dim3(HH), 0, stream>>>(colsum, lab_w1, lab_b1, lab_w2, lab_b2,
                                                   out + (long)3 * NN, NN);
}
